// MultiHeadAttention_687194767942
// MI455X (gfx1250) — matrix operand order repaired and checked by reading, not compiled, not run on hardware
//
#include <hip/hip_runtime.h>

// ---------------------------------------------------------------------------
// MHA forward for MI455X (gfx1250, wave32, WMMA 16x16x32 f16 -> f32 accum)
// B=2, H=16, S=2048, D=128.  d_out = [out (B,S,D) fp32 | weights (B,H,S,S) fp32]
// ---------------------------------------------------------------------------

#define BB 2
#define HH 16
#define SS 2048
#define DD 128

typedef __attribute__((ext_vector_type(16))) _Float16 v16h;
typedef __attribute__((ext_vector_type(8)))  _Float16 v8h;
typedef __attribute__((ext_vector_type(8)))  float    v8f;
typedef __attribute__((ext_vector_type(4)))  float    v4f;
typedef __attribute__((ext_vector_type(4)))  int      v4i;

// ---- fragment loaders (contiguous-k order k = 16*half + s, same for A and B, wave32) -------------------------
// A 16x32 f16: lane r = L&15 is row M; half = L>>4 selects K {0..15} vs {16..31}
__device__ __forceinline__ v16h load_a_frag(const _Float16* __restrict__ row, int kb, int half) {
  v16h a;
  v8h lo = *(const v8h*)(row + kb + half * 16);
  v8h hi = *(const v8h*)(row + kb + 8 + half * 16);
#pragma unroll
  for (int i = 0; i < 8; ++i) { a[i] = lo[i]; a[8 + i] = hi[i]; }
  return a;
}

// B 32x16 f16: lane n = L&15 is column N; lane reads contiguous K run of 16 at +((L>>4)*16)
__device__ __forceinline__ v16h load_b_frag(const _Float16* __restrict__ p) {
  v16h b;
  v8h lo = *(const v8h*)(p);
  v8h hi = *(const v8h*)(p + 8);
#pragma unroll
  for (int i = 0; i < 8; ++i) { b[i] = lo[i]; b[8 + i] = hi[i]; }
  return b;
}

// 16x16 tile GEMM, K=128: preload all 4 B-frags (one big clause), then 4 WMMAs
__device__ __forceinline__ v8f wmma_k128(const v16h a[4], const _Float16* __restrict__ brow) {
  v16h bf[4];
#pragma unroll
  for (int kk = 0; kk < 4; ++kk) bf[kk] = load_b_frag(brow + kk * 32);
  v8f acc = {};
#pragma unroll
  for (int kk = 0; kk < 4; ++kk)
    acc = __builtin_amdgcn_wmma_f32_16x16x32_f16(false, a[kk], false, bf[kk],
                                                 (short)0, acc, false, false);
  return acc;
}

__device__ __forceinline__ float wave_max(float v) {
#pragma unroll
  for (int m = 16; m > 0; m >>= 1) v = fmaxf(v, __shfl_xor(v, m, 32));
  return v;
}
__device__ __forceinline__ float wave_sum(float v) {
#pragma unroll
  for (int m = 16; m > 0; m >>= 1) v += __shfl_xor(v, m, 32);
  return v;
}

// ---- stage 0: fp32 -> fp16 (NT loads: source read exactly once) ------------
__global__ void cvt_f16_kernel(const float* __restrict__ src, _Float16* __restrict__ dst, int n) {
  int i = blockIdx.x * blockDim.x + threadIdx.x;
  int stride = gridDim.x * blockDim.x;
  for (; i < n; i += stride)
    dst[i] = (_Float16)__builtin_nontemporal_load(src + i);
}

// ---- stage 1: QKV projection ----------------------------------------------
// grid (B*S/16, 32), block 128.  wave -> one 16-wide column tile, shared A frags.
__global__ void __launch_bounds__(128)
qkv_kernel(const _Float16* __restrict__ xh,
           const _Float16* __restrict__ Wqh, const _Float16* __restrict__ Wkh,
           const _Float16* __restrict__ Wvh,
           const float* __restrict__ bq, const float* __restrict__ bk,
           const float* __restrict__ bv,
           _Float16* __restrict__ qh, _Float16* __restrict__ kh,
           _Float16* __restrict__ vTh) {
  const int lane = threadIdx.x & 31;
  const int wave = threadIdx.x >> 5;
  const int nl   = lane & 15;
  const int half = lane >> 4;
  const int row0 = blockIdx.x * 16;            // global row in [0, B*S)
  const int ct   = blockIdx.y * 4 + wave;      // column tile 0..127 (H*D/16)
  const int c0   = ct * 16;
  const int h    = c0 >> 7;
  const int d0   = c0 & (DD - 1);
  const int b    = row0 >> 11;
  const int sloc = row0 & (SS - 1);

  // preload A fragments (x rows, K=128)
  const _Float16* xrow = xh + (size_t)(row0 + nl) * DD;
  v16h a[4];
#pragma unroll
  for (int kk = 0; kk < 4; ++kk) a[kk] = load_a_frag(xrow, kk * 32, half);

  const size_t wro = (size_t)(c0 + nl) * DD + half * 16;
  v8f aq = wmma_k128(a, Wqh + wro);
  v8f ak = wmma_k128(a, Wkh + wro);
  v8f av = wmma_k128(a, Wvh + wro);

  const float bqn = bq[c0 + nl], bkn = bk[c0 + nl], bvn = bv[c0 + nl];

  // q,k -> (B,H,S,D) f16
#pragma unroll
  for (int i = 0; i < 8; ++i) {
    int m = i + 8 * half;
    size_t idx = (((size_t)b * HH + h) * SS + (sloc + m)) * DD + d0 + nl;
    qh[idx] = (_Float16)(aq[i] + bqn);
    kh[idx] = (_Float16)(ak[i] + bkn);
  }
  // v -> transposed (B,H,D,S) f16 : contiguous 16B store per lane
  v8h vv;
#pragma unroll
  for (int i = 0; i < 8; ++i) vv[i] = (_Float16)(av[i] + bvn);
  *(v8h*)(vTh + (((size_t)b * HH + h) * DD + d0 + nl) * SS + sloc + 8 * half) = vv;
}

// ---- stage 2: fused scores + softmax(+mask) + weights@V --------------------
// grid = B*H*(S/16) = 4096 blocks, 128 threads, 128 KB dynamic LDS (16x2048 f32)
__global__ void __launch_bounds__(128)
attn_kernel(const _Float16* __restrict__ qh, const _Float16* __restrict__ kh,
            const _Float16* __restrict__ vTh, const int* __restrict__ mask,
            float* __restrict__ weights_out, _Float16* __restrict__ ctxh) {
  extern __shared__ float sc[];                 // [16][2048]
  const int lane = threadIdx.x & 31;
  const int wave = threadIdx.x >> 5;
  const int nl   = lane & 15;
  const int half = lane >> 4;
  const int bh   = blockIdx.x >> 7;             // b*H + h
  const int s0   = (blockIdx.x & 127) * 16;
  const int b    = bh >> 4;
  const int h    = bh & 15;
  const float scale = 0.08838834764831845f;     // 1/sqrt(128)

  // ---- phase 1: scores (16 x 2048) via WMMA into LDS (mask deferred) ----
  const _Float16* qrow = qh + ((size_t)bh * SS + s0 + nl) * DD;
  v16h a[4];
#pragma unroll
  for (int kk = 0; kk < 4; ++kk) a[kk] = load_a_frag(qrow, kk * 32, half);

  for (int nt = wave; nt < SS / 16; nt += 4) {
    const int t0 = nt * 16;
    const _Float16* krow = kh + ((size_t)bh * SS + t0 + nl) * DD + half * 16;
    __builtin_prefetch(krow + 64 * DD, 0, 1);   // this wave's next tile (t0+64)
    v8f acc = wmma_k128(a, krow);
#pragma unroll
    for (int i = 0; i < 8; ++i) {
      int m = i + 8 * half;
      sc[m * SS + t0 + nl] = acc[i] * scale;
    }
  }
  __syncthreads();

  // ---- phase 2: mask + softmax (4 rows/wave), all passes 16B vectorized ----
  for (int rr = 0; rr < 4; ++rr) {
    const int r = wave * 4 + rr;
    float* row = sc + r * SS;
    const int* mrow = mask + ((size_t)b * SS + s0 + r) * SS;
    float mx = -3.4e38f;
    for (int j = lane * 4; j < SS; j += 128) {
      v4f v  = *(const v4f*)(row + j);
      v4i mk = __builtin_nontemporal_load((const v4i*)(mrow + j));  // read-once
#pragma unroll
      for (int c = 0; c < 4; ++c) {
        float t = v[c] + (float)mk[c] * (-1e9f);
        v[c] = t;
        mx = fmaxf(mx, t);
      }
      *(v4f*)(row + j) = v;
    }
    mx = wave_max(mx);
    float sum = 0.f;
    for (int j = lane * 4; j < SS; j += 128) {
      v4f v = *(const v4f*)(row + j);
#pragma unroll
      for (int c = 0; c < 4; ++c) { float e = __expf(v[c] - mx); v[c] = e; sum += e; }
      *(v4f*)(row + j) = v;
    }
    sum = wave_sum(sum);
    const float inv = 1.0f / sum;
    float* wout = weights_out + ((size_t)bh * SS + s0 + r) * SS;
    for (int j = lane * 4; j < SS; j += 128) {
      v4f v = *(const v4f*)(row + j);
#pragma unroll
      for (int c = 0; c < 4; ++c) v[c] *= inv;
      *(v4f*)(row + j) = v;                                   // f32 weights for phase 3
      __builtin_nontemporal_store(v, (v4f*)(wout + j));       // 537 MB streaming store (NT)
    }
  }
  __syncthreads();

  // ---- phase 3: ctx = weights @ V (K=2048), double-buffered B frags ----
  v8f acc0 = {}, acc1 = {};
  const int d0a = wave * 32, d0b = wave * 32 + 16;
  const _Float16* vrow0 = vTh + ((size_t)bh * DD + d0a + nl) * SS + half * 16;
  const _Float16* vrow1 = vTh + ((size_t)bh * DD + d0b + nl) * SS + half * 16;
  const float* wrow = sc + nl * SS + half * 16;

  v16h b0 = load_b_frag(vrow0);
  v16h b1 = load_b_frag(vrow1);
  for (int kt = 0; kt < SS / 32; ++kt) {
    const int kb = kt * 32;
    const int kn = (kt < SS / 32 - 1) ? kb + 32 : kb;
    v4f f0 = *(const v4f*)(wrow + kb);
    v4f f1 = *(const v4f*)(wrow + kb + 4);
    v4f f2 = *(const v4f*)(wrow + kb + 8);
    v4f f3 = *(const v4f*)(wrow + kb + 12);
    v16h af;
#pragma unroll
    for (int i = 0; i < 4; ++i) {
      af[i]      = (_Float16)f0[i];
      af[4 + i]  = (_Float16)f1[i];
      af[8 + i]  = (_Float16)f2[i];
      af[12 + i] = (_Float16)f3[i];
    }
    v16h nb0 = load_b_frag(vrow0 + kn);   // next-iteration loads issue before WMMAs
    v16h nb1 = load_b_frag(vrow1 + kn);
    acc0 = __builtin_amdgcn_wmma_f32_16x16x32_f16(false, af, false, b0, (short)0, acc0, false, false);
    acc1 = __builtin_amdgcn_wmma_f32_16x16x32_f16(false, af, false, b1, (short)0, acc1, false, false);
    b0 = nb0; b1 = nb1;
  }
  // ctx -> (B, S, H*D) f16
#pragma unroll
  for (int i = 0; i < 8; ++i) {
    int m = i + 8 * half;
    size_t base = ((size_t)b * SS + s0 + m) * (HH * DD) + h * DD;
    ctxh[base + d0a + nl] = (_Float16)acc0[i];
    ctxh[base + d0b + nl] = (_Float16)acc1[i];
  }
}

// ---- stage 3: out = ctx @ Wd^T + bd ---------------------------------------
// grid = B*S/16 blocks, 128 threads; wave handles 2 of 8 column tiles.
__global__ void __launch_bounds__(128)
proj_kernel(const _Float16* __restrict__ ctxh, const _Float16* __restrict__ Wdh,
            const float* __restrict__ bd, float* __restrict__ out) {
  const int lane = threadIdx.x & 31;
  const int wave = threadIdx.x >> 5;
  const int nl   = lane & 15;
  const int half = lane >> 4;
  const int row0 = blockIdx.x * 16;
  const int e0a  = wave * 32, e0b = wave * 32 + 16;

  const _Float16* arow  = ctxh + (size_t)(row0 + nl) * (HH * DD);
  const _Float16* b0row = Wdh + (size_t)(e0a + nl) * (HH * DD) + half * 16;
  const _Float16* b1row = Wdh + (size_t)(e0b + nl) * (HH * DD) + half * 16;

  v8f acc0 = {}, acc1 = {};
  v16h b0 = load_b_frag(b0row);
  v16h b1 = load_b_frag(b1row);
  for (int kt = 0; kt < (HH * DD) / 32; ++kt) {
    const int kb = kt * 32;
    const int kn = (kt < (HH * DD) / 32 - 1) ? kb + 32 : kb;
    __builtin_prefetch(arow + kb + 256, 0, 1);
    v16h af  = load_a_frag(arow, kb, half);
    v16h nb0 = load_b_frag(b0row + kn);
    v16h nb1 = load_b_frag(b1row + kn);
    acc0 = __builtin_amdgcn_wmma_f32_16x16x32_f16(false, af, false, b0, (short)0, acc0, false, false);
    acc1 = __builtin_amdgcn_wmma_f32_16x16x32_f16(false, af, false, b1, (short)0, acc1, false, false);
    b0 = nb0; b1 = nb1;
  }
  const float bd0 = bd[e0a + nl], bd1 = bd[e0b + nl];
#pragma unroll
  for (int i = 0; i < 8; ++i) {
    int m = i + 8 * half;
    __builtin_nontemporal_store(acc0[i] + bd0, out + (size_t)(row0 + m) * DD + e0a + nl);
    __builtin_nontemporal_store(acc1[i] + bd1, out + (size_t)(row0 + m) * DD + e0b + nl);
  }
}

// ---------------------------------------------------------------------------
extern "C" void kernel_launch(void* const* d_in, const int* in_sizes, int n_in,
                              void* d_out, int out_size, void* d_ws, size_t ws_size,
                              hipStream_t stream) {
  (void)in_sizes; (void)n_in; (void)out_size; (void)ws_size;
  const float* x    = (const float*)d_in[0];
  const int*   mask = (const int*)d_in[1];
  const float* Wq   = (const float*)d_in[2];
  const float* bq   = (const float*)d_in[3];
  const float* Wk   = (const float*)d_in[4];
  const float* bk   = (const float*)d_in[5];
  const float* Wv   = (const float*)d_in[6];
  const float* bv   = (const float*)d_in[7];
  const float* Wd   = (const float*)d_in[8];
  const float* bd   = (const float*)d_in[9];

  float* out     = (float*)d_out;
  float* weights = out + (size_t)BB * SS * DD;   // tuple: (out, weights) flat

  // workspace carve-up (all offsets 256B-aligned)
  char* ws = (char*)d_ws;
  _Float16* xh   = (_Float16*)(ws + 0);           //  1,048,576 B
  _Float16* Wqh  = (_Float16*)(ws + 1048576);     //    524,288 B
  _Float16* Wkh  = (_Float16*)(ws + 1572864);
  _Float16* Wvh  = (_Float16*)(ws + 2097152);
  _Float16* Wdh  = (_Float16*)(ws + 2621440);
  _Float16* qh   = (_Float16*)(ws + 3145728);     // 16,777,216 B
  _Float16* kh   = (_Float16*)(ws + 19922944);
  _Float16* vTh  = (_Float16*)(ws + 36700160);
  _Float16* ctxh = (_Float16*)(ws + 53477376);    // end ~70 MB

  // stage 0: conversions
  cvt_f16_kernel<<<512, 256, 0, stream>>>(x,  xh,  BB * SS * DD);
  cvt_f16_kernel<<<256, 256, 0, stream>>>(Wq, Wqh, HH * DD * DD);
  cvt_f16_kernel<<<256, 256, 0, stream>>>(Wk, Wkh, HH * DD * DD);
  cvt_f16_kernel<<<256, 256, 0, stream>>>(Wv, Wvh, HH * DD * DD);
  cvt_f16_kernel<<<256, 256, 0, stream>>>(Wd, Wdh, HH * DD * DD);

  // stage 1: QKV projection
  qkv_kernel<<<dim3(BB * SS / 16, 32), 128, 0, stream>>>(
      xh, Wqh, Wkh, Wvh, bq, bk, bv, qh, kh, vTh);

  // stage 2: fused attention (128 KB dynamic LDS; WGP has 320 KB)
  hipFuncSetAttribute((const void*)attn_kernel,
                      hipFuncAttributeMaxDynamicSharedMemorySize, 131072);
  attn_kernel<<<BB * HH * (SS / 16), 128, 131072, stream>>>(
      qh, kh, vTh, mask, weights, ctxh);

  // stage 3: output projection
  proj_kernel<<<BB * SS / 16, 128, 0, stream>>>(ctxh, Wdh, bd, out);
}
